// CausalSelfAttention_34411277976012
// MI455X (gfx1250) — matrix operand order repaired and checked by reading, not compiled, not run on hardware
//
#include <hip/hip_runtime.h>

// ---------- CDNA5 (gfx1250) wave32 WMMA types ----------
typedef __attribute__((ext_vector_type(16))) __bf16 v16bf;
typedef __attribute__((ext_vector_type(8)))  __bf16 v8bf;
typedef __attribute__((ext_vector_type(8)))  float  v8f;
typedef unsigned int u32;
typedef __attribute__((ext_vector_type(4))) u32 u32x4;
typedef __attribute__((ext_vector_type(8))) int i32x8;
typedef __attribute__((ext_vector_type(4))) int i32x4;

#define WMMA_BF16(A, B, C) \
  __builtin_amdgcn_wmma_f32_16x16x32_bf16(false, (A), false, (B), (short)0, (C), false, false)

__device__ __forceinline__ v16bf make_frag(const __bf16* p0, const __bf16* p1) {
  union { v16bf v; v8bf h[2]; } u;
  u.h[0] = *(const v8bf*)p0;   // elements 0..7
  u.h[1] = *(const v8bf*)p1;   // elements 8..15
  return u.v;
}

// ---------- Tensor Data Mover: 2D bf16 tile load (D# per CDNA5 ISA) ----------
// tile_d0 = row width (2B elems), tile_d1 = rows, stride0 = row stride (2B elems).
// This toolchain (clang-23 / therock-10.0) exposes the 6-arg builtin form.
__device__ __forceinline__ void tdm_load_2d_bf16(unsigned lds_byte, const void* gsrc,
                                                 unsigned tile_d0, unsigned tile_d1,
                                                 unsigned long long stride0) {
  unsigned long long ga = (unsigned long long)gsrc;
  u32x4 g0;
  g0[0] = 1u;                                        // count=1, user mode, no gather
  g0[1] = lds_byte;                                  // lds_addr
  g0[2] = (u32)ga;                                   // global_addr[31:0]
  g0[3] = (u32)((ga >> 32) & 0x01ffffffu) | (2u << 30);  // addr[56:32] | type=2
  i32x8 g1;
  g1[0] = (int)(1u << 16);                           // data_size=1 (2 bytes); wg_mask=0
  g1[1] = (int)((tile_d0 & 0xffffu) << 16);          // tensor_dim0[15:0]
  g1[2] = (int)(((tile_d0 >> 16) & 0xffffu) | ((tile_d1 & 0xffffu) << 16)); // dim0 hi | dim1 lo
  g1[3] = (int)(((tile_d1 >> 16) & 0xffffu) | ((tile_d0 & 0xffffu) << 16)); // dim1 hi | tile_dim0
  g1[4] = (int)(tile_d1 & 0xffffu);                  // tile_dim1; tile_dim2=0
  g1[5] = (int)(u32)stride0;                         // tensor_dim0_stride[31:0]
  g1[6] = (int)((stride0 >> 32) & 0xffffu);          // stride0[47:32]; stride1=0
  g1[7] = 0;
  i32x4 z4 = {};                                     // groups 2/3 unused (<=2D tiles)
  i32x8 z8 = {};                                     // trailing descriptor group (unused)
  __builtin_amdgcn_tensor_load_to_lds(g0, g1, z4, z4, z8, 0);
}

// Problem constants
#define BB 4
#define TT 2048
#define CC 1024
#define HH 16
#define DH 64

// ---------- fp32 -> bf16 copy ----------
__global__ void conv_bf16_kernel(const float* __restrict__ in, __bf16* __restrict__ out, int n) {
  int i = blockIdx.x * 256 + threadIdx.x;
  if (i < n) out[i] = (__bf16)in[i];
}

// in [rows][cols] fp32 -> out [cols][rows] bf16  (N-major weights for WMMA B-frags)
__global__ void transpose_bf16_kernel(const float* __restrict__ in, __bf16* __restrict__ out,
                                      int rows, int cols) {
  int i = blockIdx.x * 256 + threadIdx.x;
  if (i < rows * cols) {
    int r = i / cols, c = i - r * cols;
    out[(size_t)c * rows + r] = (__bf16)in[i];
  }
}

// ---------- QKV GEMM: A[8192][1024] x BT[3072][1024] -> scatter q,k,v^T ----------
__global__ void gemm_qkv_kernel(const __bf16* __restrict__ A, const __bf16* __restrict__ BT,
                                __bf16* __restrict__ qb, __bf16* __restrict__ kb,
                                __bf16* __restrict__ vtb) {
  const int lane = threadIdx.x & 31, w = threadIdx.x >> 5;
  const int lanelo = lane & 15, half = lane >> 4;
  const int wm = w & 1, wn = w >> 1;
  const int m0 = blockIdx.y * 64 + wm * 32;
  const int n0 = blockIdx.x * 256 + wn * 64;
  const int K = CC;

  v8f acc[2][4] = {};
  for (int kk = 0; kk < K; kk += 32) {
    const __bf16* ap0 = A + (size_t)(m0 + lanelo) * K + kk;
    const __bf16* ap1 = A + (size_t)(m0 + 16 + lanelo) * K + kk;
    __builtin_prefetch(ap0 + 64, 0, 3);   // near-scope prefetch of next K-chunk
    v16bf a0 = make_frag(ap0 + half * 8, ap0 + 16 + half * 8);
    v16bf a1 = make_frag(ap1 + half * 8, ap1 + 16 + half * 8);
#pragma unroll
    for (int nt = 0; nt < 4; ++nt) {
      const __bf16* bp = BT + (size_t)(n0 + nt * 16 + lanelo) * K + kk + half * 8;
      v16bf b = make_frag(bp, bp + 16);
      acc[0][nt] = WMMA_BF16(a0, b, acc[0][nt]);
      acc[1][nt] = WMMA_BF16(a1, b, acc[1][nt]);
    }
  }
#pragma unroll
  for (int mt = 0; mt < 2; ++mt)
#pragma unroll
    for (int nt = 0; nt < 4; ++nt)
#pragma unroll
      for (int r = 0; r < 8; ++r) {
        int m = m0 + mt * 16 + r + 8 * half;
        int n = n0 + nt * 16 + lanelo;
        int bidx = m >> 11, t = m & (TT - 1);
        int which = n >> 10, hn = n & (CC - 1);
        int h = hn >> 6, d = hn & (DH - 1);
        __bf16 bv = (__bf16)acc[mt][nt][r];
        size_t bh = (size_t)bidx * HH + h;
        if (which == 0)      qb[(bh * TT + t) * DH + d] = bv;
        else if (which == 1) kb[(bh * TT + t) * DH + d] = bv;
        else                 vtb[(bh * DH + d) * TT + t] = bv;
      }
}

// ---------- Flash attention with TDM-staged K/V tiles ----------
// grid (T/128, B*H), 8 waves; wave w owns q-rows [q0w, q0w+16).
// Double-buffered LDS K/V tiles DMA'd by the Tensor Data Mover (wave 0 issues).
__global__ void attn_kernel(const __bf16* __restrict__ Q, const __bf16* __restrict__ Km,
                            const __bf16* __restrict__ VT, __bf16* __restrict__ Y) {
  // [ buf0: K(64x64) V(64x64) | buf1: K V | P tiles 8x(16x64) ] = 48 KB
  __shared__ __bf16 smem[2 * 8192 + 8 * 1024];
  const int lane = threadIdx.x & 31, w = threadIdx.x >> 5;
  const int lanelo = lane & 15, half = lane >> 4;
  const int qblk = blockIdx.x, bh = blockIdx.y;
  const int b = bh >> 4, h = bh & (HH - 1);
  const int q0w = qblk * 128 + w * 16;
  const int qmax_w = q0w + 15;
  const int qmax_blk = qblk * 128 + 127;

  const __bf16* qbase  = Q  + (size_t)bh * TT * DH;
  const __bf16* kbase  = Km + (size_t)bh * TT * DH;
  const __bf16* vtbase = VT + (size_t)bh * DH * TT;
  __bf16* lp = smem + 2 * 8192 + w * 1024;   // this wave's P tile

  // Q A-fragments, kept in registers for the whole KV loop
  const __bf16* qp = qbase + (size_t)(q0w + lanelo) * DH;
  v16bf aQ0 = make_frag(qp + half * 8,      qp + 16 + half * 8);
  v16bf aQ1 = make_frag(qp + 32 + half * 8, qp + 48 + half * 8);

  v8f o[4] = {};
  float mrun[8], lrun[8];
#pragma unroll
  for (int r = 0; r < 8; ++r) { mrun[r] = -__builtin_inff(); lrun[r] = 0.f; }

  // Prologue: DMA tile j0=0 into buffer 0 (K: contiguous 8KB; V^T: 64 rows x 128B, stride 4KB)
  if (w == 0) {
    unsigned base0 = (unsigned)(size_t)(const void*)smem;
    tdm_load_2d_bf16(base0,            kbase,      4096, 1, 4096);
    tdm_load_2d_bf16(base0 + 8192,     vtbase,     64,  64, (unsigned long long)TT);
  }

  int buf = 0;
  for (int j0 = 0; j0 <= qmax_blk; j0 += 64) {
    // Issue next tile's DMA into the other buffer, then wait for current tile.
    if (w == 0) {
      if (j0 + 64 <= qmax_blk) {
        unsigned basen = (unsigned)(size_t)(const void*)(smem + (buf ^ 1) * 8192);
        tdm_load_2d_bf16(basen,        kbase + (size_t)(j0 + 64) * DH, 4096, 1, 4096);
        tdm_load_2d_bf16(basen + 8192, vtbase + (j0 + 64),             64,  64,
                         (unsigned long long)TT);
        __builtin_amdgcn_s_wait_tensorcnt(2);   // 2 newest outstanding; current tile done
      } else {
        __builtin_amdgcn_s_wait_tensorcnt(0);
      }
    }
    __syncthreads();

    if (j0 <= qmax_w) {
      const __bf16* ldsK = smem + buf * 8192;
      const __bf16* ldsV = ldsK + 4096;
      // S = Q K^T (8 WMMAs), K frags from LDS
      v8f s[4];
#pragma unroll
      for (int nt = 0; nt < 4; ++nt) {
        const __bf16* kp = ldsK + (nt * 16 + lanelo) * 64 + half * 8;
        v16bf b0 = make_frag(kp, kp + 16);
        v16bf b1 = make_frag(kp + 32, kp + 48);
        v8f z = {};
        z = WMMA_BF16(aQ0, b0, z);
        z = WMMA_BF16(aQ1, b1, z);
        s[nt] = z;
      }
      // scale + causal mask
#pragma unroll
      for (int nt = 0; nt < 4; ++nt)
#pragma unroll
        for (int r = 0; r < 8; ++r) {
          int row = q0w + r + 8 * half;
          int col = j0 + nt * 16 + lanelo;
          float v = s[nt][r] * 0.125f;
          s[nt][r] = (col > row) ? -__builtin_inff() : v;
        }
      // online softmax; rows live in 16-lane half-wave groups -> shfl_xor masks 8..1
      float pv[4][8];
#pragma unroll
      for (int r = 0; r < 8; ++r) {
        float rv = fmaxf(fmaxf(s[0][r], s[1][r]), fmaxf(s[2][r], s[3][r]));
        rv = fmaxf(rv, __shfl_xor(rv, 8, 32));
        rv = fmaxf(rv, __shfl_xor(rv, 4, 32));
        rv = fmaxf(rv, __shfl_xor(rv, 2, 32));
        rv = fmaxf(rv, __shfl_xor(rv, 1, 32));
        float mnew = fmaxf(mrun[r], rv);
        float esc  = __expf(mrun[r] - mnew);
        float ps = 0.f;
#pragma unroll
        for (int nt = 0; nt < 4; ++nt) { float p = __expf(s[nt][r] - mnew); pv[nt][r] = p; ps += p; }
        ps += __shfl_xor(ps, 8, 32);
        ps += __shfl_xor(ps, 4, 32);
        ps += __shfl_xor(ps, 2, 32);
        ps += __shfl_xor(ps, 1, 32);
        lrun[r] = lrun[r] * esc + ps;
        mrun[r] = mnew;
#pragma unroll
        for (int nt = 0; nt < 4; ++nt) o[nt][r] *= esc;
      }
      // C/D layout -> A-fragment layout via wave-private LDS P tile
#pragma unroll
      for (int nt = 0; nt < 4; ++nt)
#pragma unroll
        for (int r = 0; r < 8; ++r)
          lp[(r + 8 * half) * 64 + nt * 16 + lanelo] = (__bf16)pv[nt][r];
      asm volatile("s_wait_dscnt 0" ::: "memory");
      const __bf16* pr = lp + lanelo * 64;
      v16bf aP0 = make_frag(pr + half * 8,      pr + 16 + half * 8);
      v16bf aP1 = make_frag(pr + 32 + half * 8, pr + 48 + half * 8);
      // O += P V (8 WMMAs), V^T frags from LDS
#pragma unroll
      for (int nt = 0; nt < 4; ++nt) {
        const __bf16* vp = ldsV + (nt * 16 + lanelo) * 64 + half * 8;
        v16bf b0 = make_frag(vp, vp + 16);
        v16bf b1 = make_frag(vp + 32, vp + 48);
        o[nt] = WMMA_BF16(aP0, b0, o[nt]);
        o[nt] = WMMA_BF16(aP1, b1, o[nt]);
      }
    }
    __syncthreads();   // all readers done before next DMA overwrites this buffer
    buf ^= 1;
  }
  // normalize and write y (bf16, [B,T,C] with head-interleaved columns)
#pragma unroll
  for (int r = 0; r < 8; ++r) {
    float inv = 1.f / lrun[r];
    int row = q0w + r + 8 * half;
#pragma unroll
    for (int nt = 0; nt < 4; ++nt)
      Y[((size_t)b * TT + row) * CC + h * DH + nt * 16 + lanelo] = (__bf16)(o[nt][r] * inv);
  }
}

// ---------- Output projection: A[8192][1024] x BT[1024][1024] -> fp32 out ----------
__global__ void gemm_out_kernel(const __bf16* __restrict__ A, const __bf16* __restrict__ BT,
                                float* __restrict__ out) {
  const int lane = threadIdx.x & 31, w = threadIdx.x >> 5;
  const int lanelo = lane & 15, half = lane >> 4;
  const int wm = w & 1, wn = w >> 1;
  const int m0 = blockIdx.y * 64 + wm * 32;
  const int n0 = blockIdx.x * 256 + wn * 64;
  const int K = CC;

  v8f acc[2][4] = {};
  for (int kk = 0; kk < K; kk += 32) {
    const __bf16* ap0 = A + (size_t)(m0 + lanelo) * K + kk;
    const __bf16* ap1 = A + (size_t)(m0 + 16 + lanelo) * K + kk;
    __builtin_prefetch(ap0 + 64, 0, 3);
    v16bf a0 = make_frag(ap0 + half * 8, ap0 + 16 + half * 8);
    v16bf a1 = make_frag(ap1 + half * 8, ap1 + 16 + half * 8);
#pragma unroll
    for (int nt = 0; nt < 4; ++nt) {
      const __bf16* bp = BT + (size_t)(n0 + nt * 16 + lanelo) * K + kk + half * 8;
      v16bf b = make_frag(bp, bp + 16);
      acc[0][nt] = WMMA_BF16(a0, b, acc[0][nt]);
      acc[1][nt] = WMMA_BF16(a1, b, acc[1][nt]);
    }
  }
#pragma unroll
  for (int mt = 0; mt < 2; ++mt)
#pragma unroll
    for (int nt = 0; nt < 4; ++nt)
#pragma unroll
      for (int r = 0; r < 8; ++r) {
        int m = m0 + mt * 16 + r + 8 * half;
        int n = n0 + nt * 16 + lanelo;
        out[(size_t)m * CC + n] = acc[mt][nt][r];
      }
}

// ---------- host launcher ----------
extern "C" void kernel_launch(void* const* d_in, const int* in_sizes, int n_in,
                              void* d_out, int out_size, void* d_ws, size_t ws_size,
                              hipStream_t stream) {
  const float* x    = (const float*)d_in[0];   // [4,2048,1024]
  const float* wqkv = (const float*)d_in[1];   // [1024,3072]
  const float* wout = (const float*)d_in[2];   // [1024,1024]
  float* out = (float*)d_out;                  // [4,2048,1024] fp32

  const size_t MTOT = (size_t)BB * TT;         // 8192
  char* ws = (char*)d_ws;
  __bf16* xb    = (__bf16*)ws;                 ws += MTOT * CC * 2;           // 16 MB (reused as y)
  __bf16* wqkvT = (__bf16*)ws;                 ws += (size_t)3 * CC * CC * 2; // 6 MB
  __bf16* woutT = (__bf16*)ws;                 ws += (size_t)CC * CC * 2;     // 2 MB
  __bf16* qb    = (__bf16*)ws;                 ws += MTOT * CC * 2;           // 16 MB
  __bf16* kb    = (__bf16*)ws;                 ws += MTOT * CC * 2;           // 16 MB
  __bf16* vtb   = (__bf16*)ws;                 ws += MTOT * CC * 2;           // 16 MB
  __bf16* yb    = xb;  // x-bf16 is dead after the QKV GEMM; reuse for y

  int nx = (int)(MTOT * CC);
  conv_bf16_kernel<<<nx / 256, 256, 0, stream>>>(x, xb, nx);
  transpose_bf16_kernel<<<(3 * CC * CC) / 256, 256, 0, stream>>>(wqkv, wqkvT, CC, 3 * CC);
  transpose_bf16_kernel<<<(CC * CC) / 256, 256, 0, stream>>>(wout, woutT, CC, CC);

  gemm_qkv_kernel<<<dim3(12, 128), 256, 0, stream>>>(xb, wqkvT, qb, kb, vtb);
  attn_kernel<<<dim3(TT / 128, BB * HH), 256, 0, stream>>>(qb, kb, vtb, yb);
  gemm_out_kernel<<<dim3(4, 128), 256, 0, stream>>>(yb, woutT, out);
}
